// Titan_14568529068553
// MI455X (gfx1250) — matrix operand order repaired and checked by reading, not compiled, not run on hardware
//
#include <hip/hip_runtime.h>
#include <hip/hip_bf16.h>
#include <cstdint>

// ---------------------------------------------------------------------------
// CDNA5 / gfx1250 bf16 WMMA transformer forward pass.
// Compute-bound (~17 dense TFLOP) -> all large GEMMs + attention run on
// v_wmma_f32_16x16x32_bf16. Residual stream & softmax stats stay fp32.
// GEMM uses a double-buffered LDS pipeline; A tiles go through the CDNA5
// async global->LDS path (ASYNCcnt) when the builtin is available.
// ---------------------------------------------------------------------------

typedef __attribute__((ext_vector_type(16))) __bf16 v16bf;
typedef __attribute__((ext_vector_type(8)))  float  v8f;
typedef __attribute__((ext_vector_type(4)))  int    v4i;
typedef unsigned short u16;

// address-space-qualified views for the async global->LDS builtin
typedef __attribute__((address_space(1))) v4i gv4i;
typedef __attribute__((address_space(3))) v4i lv4i;

union Frag8 { uint32_t u[8]; v16bf v; };

#if __has_builtin(__builtin_amdgcn_global_load_async_to_lds_b128)
#define HAS_ASYNC_LDS 1
#else
#define HAS_ASYNC_LDS 0
#endif

__device__ __forceinline__ void async_copy16(const u16* g, u16* l) {
#if HAS_ASYNC_LDS
  __builtin_amdgcn_global_load_async_to_lds_b128(
      (gv4i*)(uintptr_t)g, (lv4i*)(uintptr_t)l, 0, 0);
#else
  uint4 t = *(const uint4*)g;
  *(uint4*)l = t;
#endif
}
__device__ __forceinline__ void async_wait0() {
#if HAS_ASYNC_LDS && __has_builtin(__builtin_amdgcn_s_wait_asynccnt)
  __builtin_amdgcn_s_wait_asynccnt(0);
#elif HAS_ASYNC_LDS
  asm volatile("s_wait_asynccnt 0" ::: "memory");
#endif
}

__device__ __forceinline__ u16 f2bf(float f) {
  uint32_t u = __float_as_uint(f);
  u += 0x7fffu + ((u >> 16) & 1u);            // round-to-nearest-even
  return (u16)(u >> 16);
}
__device__ __forceinline__ float bf2f(u16 h) {
  return __uint_as_float(((uint32_t)h) << 16);
}
__device__ __forceinline__ float gelu_exact(float x) {
  return 0.5f * x * (1.0f + erff(x * 0.70710678118654752f));
}
// 16-bit A-matrix 16x32 fragment K offsets (ISA 7.12.2): lanes 0-15 hold
// K 0..7 & 16..23, lanes 16-31 hold K 8..15 & 24..31, packed in K-pairs.
__device__ __forceinline__ int a_koff(int v, int hi) {
  return (v < 4 ? 2 * v : 8 + 2 * v) + (hi ? 8 : 0);
}

enum { EPI_BF16 = 0, EPI_BF16_GELU = 1, EPI_RESID = 2, EPI_F32 = 3 };

// ---------------------------------------------------------------------------
// Generic bf16 GEMM: C = A(MxK,row) * B(KxN,row) + bias, with epilogue.
// 256 threads = 8 waves; block tile 128x128; BK=32 (one WMMA K-step).
// Double-buffered LDS; one barrier per K-step; A tile staged asynchronously.
// ---------------------------------------------------------------------------
template <int EPI>
__global__ __launch_bounds__(256) void gemm_bf16_kernel(
    const u16* __restrict__ A, const u16* __restrict__ B,
    const float* __restrict__ bias, void* __restrict__ out,
    float* __restrict__ resid, int M, int N, int K)
{
  __shared__ u16 As[2][128][40];   // [m][k], padded stride (80B, 16B aligned)
  __shared__ u16 Bs[2][128][40];   // [n][k] (B staged transposed)

  const int bm = blockIdx.y * 128;
  const int bn = blockIdx.x * 128;
  const int tid  = threadIdx.x;
  const int lane = tid & 31;
  const int wid  = tid >> 5;
  const int wm = (wid & 3) * 32;      // 4 wave-rows
  const int wn = (wid >> 2) * 64;     // 2 wave-cols
  const int n16 = lane & 15;
  const int hi  = lane >> 4;

  const v8f vzero = {0.f,0.f,0.f,0.f,0.f,0.f,0.f,0.f};
  v8f acc[2][4];
#pragma unroll
  for (int r = 0; r < 2; ++r)
#pragma unroll
    for (int c = 0; c < 4; ++c) acc[r][c] = vzero;

  const int ar = tid >> 1, ac = (tid & 1) << 4;     // A stage: 128 rows x 32
  const int bk = tid >> 3, bnc = (tid & 7) << 4;    // B stage: 32 rows x 128

  union BStage { uint4 q[2]; u16 s[16]; };
  BStage st;

  // --- tile 0 prologue ---
  {
    const u16* asrc = A + (size_t)(bm + ar) * K + ac;
    async_copy16(asrc,     &As[0][ar][ac]);
    async_copy16(asrc + 8, &As[0][ar][ac + 8]);
    const u16* bsrc = B + (size_t)bk * N + bn + bnc;
    st.q[0] = *(const uint4*)bsrc;             // guarded at store (N-padded ws)
    st.q[1] = *(const uint4*)(bsrc + 8);
#pragma unroll
    for (int i = 0; i < 16; ++i)
      Bs[0][bnc + i][bk] = (bn + bnc + i < N) ? st.s[i] : (u16)0;
    async_wait0();
  }
  __syncthreads();

  int buf = 0;
  for (int k0 = 0; k0 < K; k0 += 32) {
    const bool nxt = (k0 + 32) < K;
    if (nxt) {
      // issue next A tile async + next B tile global loads (latency hidden
      // under the WMMA block below)
      const u16* asrc = A + (size_t)(bm + ar) * K + k0 + 32 + ac;
      async_copy16(asrc,     &As[buf ^ 1][ar][ac]);
      async_copy16(asrc + 8, &As[buf ^ 1][ar][ac + 8]);
      const u16* bsrc = B + (size_t)(k0 + 32 + bk) * N + bn + bnc;
      st.q[0] = *(const uint4*)bsrc;
      st.q[1] = *(const uint4*)(bsrc + 8);
    }

    Frag8 af[2], bf[4];
#pragma unroll
    for (int r = 0; r < 2; ++r)
#pragma unroll
      for (int v = 0; v < 8; ++v)
        af[r].u[v] = *(const uint32_t*)&As[buf][wm + r * 16 + n16][a_koff(v, hi)];
#pragma unroll
    for (int c = 0; c < 4; ++c)
#pragma unroll
      for (int v = 0; v < 8; ++v)
        bf[c].u[v] = *(const uint32_t*)&Bs[buf][wn + c * 16 + n16][a_koff(v, hi)];

#pragma unroll
    for (int r = 0; r < 2; ++r)
#pragma unroll
      for (int c = 0; c < 4; ++c)
        acc[r][c] = __builtin_amdgcn_wmma_f32_16x16x32_bf16(
            false, af[r].v, false, bf[c].v, (short)0, acc[r][c], false, false);

    if (nxt) {
#pragma unroll
      for (int i = 0; i < 16; ++i)
        Bs[buf ^ 1][bnc + i][bk] = (bn + bnc + i < N) ? st.s[i] : (u16)0;
      async_wait0();
    }
    __syncthreads();
    buf ^= 1;
  }

  // --- epilogue: C layout = VGPR e -> M = e + hi*8, lane n16 -> N ---
#pragma unroll
  for (int r = 0; r < 2; ++r)
#pragma unroll
    for (int c = 0; c < 4; ++c)
#pragma unroll
      for (int e = 0; e < 8; ++e) {
        int grow = bm + wm + r * 16 + e + hi * 8;
        int gcol = bn + wn + c * 16 + n16;
        if (gcol >= N || grow >= M) continue;
        float val = acc[r][c][e] + bias[gcol];
        size_t idx = (size_t)grow * N + gcol;
        if (EPI == EPI_RESID)            resid[idx] += val;
        else if (EPI == EPI_BF16)        ((u16*)out)[idx] = f2bf(val);
        else if (EPI == EPI_BF16_GELU)   ((u16*)out)[idx] = f2bf(gelu_exact(val));
        else                             ((float*)out)[idx] = val;
      }
}

// ---------------------------------------------------------------------------
// Flash attention: one wave per (batch, head, 16-query tile). d_k = 64.
// ---------------------------------------------------------------------------
__global__ __launch_bounds__(32) void attn_kernel(
    const u16* __restrict__ qb, const u16* __restrict__ kb,
    const u16* __restrict__ vb, const float* __restrict__ btab,
    u16* __restrict__ ctxb)
{
  const int qt   = blockIdx.x;     // 0..4 (covers 80 slots, 65 valid)
  const int head = blockIdx.y;     // 0..15
  const int b    = blockIdx.z;     // 0..511
  const int lane = threadIdx.x;
  const int n16  = lane & 15;
  const int hi   = lane >> 4;
  const size_t base = (size_t)b * 65 * 1024 + (size_t)head * 64;

  __shared__ u16 Pl[16][20];

  const v8f vzero = {0.f,0.f,0.f,0.f,0.f,0.f,0.f,0.f};

  // Q fragments (A-matrix, 16 queries x 64 features -> 2 K-chunks)
  Frag8 qf[2];
  {
    int t = qt * 16 + n16; int tc = t < 65 ? t : 64;
    const u16* qrow = qb + base + (size_t)tc * 1024;
#pragma unroll
    for (int c = 0; c < 2; ++c)
#pragma unroll
      for (int v = 0; v < 8; ++v)
        qf[c].u[v] = *(const uint32_t*)(qrow + c * 32 + a_koff(v, hi));
  }

  float mold[8], lsum[8];
  v8f oacc[4];
#pragma unroll
  for (int c = 0; c < 4; ++c) oacc[c] = vzero;
#pragma unroll
  for (int e = 0; e < 8; ++e) { mold[e] = -1e30f; lsum[e] = 0.f; }

  for (int kt = 0; kt < 5; ++kt) {
    // K^T fragments (B-matrix: features x 16 keys)
    Frag8 kf[2];
    {
      int u = kt * 16 + n16; int uc = u < 65 ? u : 64;
      const u16* krow = kb + base + (size_t)uc * 1024;
#pragma unroll
      for (int c = 0; c < 2; ++c)
#pragma unroll
        for (int v = 0; v < 8; ++v)
          kf[c].u[v] = *(const uint32_t*)(krow + c * 32 + a_koff(v, hi));
    }
    v8f sc = vzero;
    sc = __builtin_amdgcn_wmma_f32_16x16x32_bf16(false, qf[0].v, false, kf[0].v, (short)0, sc, false, false);
    sc = __builtin_amdgcn_wmma_f32_16x16x32_bf16(false, qf[1].v, false, kf[1].v, (short)0, sc, false, false);

    float pval[8];
#pragma unroll
    for (int e = 0; e < 8; ++e) {
      int t = qt * 16 + e + hi * 8;     // query token
      int u = kt * 16 + n16;            // key token
      float s;
      if (u >= 65) s = -1e30f;
      else {
        s = sc[e] * 0.125f;             // 1/sqrt(64)
        if (t >= 1 && t < 65 && u >= 1) {   // relative-position bias (board only)
          int p = t - 1, q = u - 1;
          int dr = (p >> 3) - (q >> 3); dr = dr < -7 ? -7 : (dr > 7 ? 7 : dr);
          int df = (p & 7) - (q & 7);   df = df < -7 ? -7 : (df > 7 ? 7 : df);
          s += btab[((dr + 7) * 15 + (df + 7)) * 16 + head];
        }
      }
      // online softmax: row stats shared across the 16 lanes of this half
      float mt = s;
#pragma unroll
      for (int msk = 1; msk < 16; msk <<= 1) mt = fmaxf(mt, __shfl_xor(mt, msk, 32));
      float mnew  = fmaxf(mold[e], mt);
      float alpha = __expf(mold[e] - mnew);
      float pe    = __expf(s - mnew);
      float ps    = pe;
#pragma unroll
      for (int msk = 1; msk < 16; msk <<= 1) ps += __shfl_xor(ps, msk, 32);
      lsum[e] = lsum[e] * alpha + ps;
      mold[e] = mnew;
      pval[e] = pe;
#pragma unroll
      for (int c = 0; c < 4; ++c) oacc[c][e] *= alpha;
    }

    // transpose P (C layout) -> A-fragment layout via LDS, K padded to 32
    __syncthreads();
#pragma unroll
    for (int e = 0; e < 8; ++e) Pl[e + hi * 8][n16] = f2bf(pval[e]);
    __syncthreads();
    Frag8 pf;
#pragma unroll
    for (int v = 0; v < 8; ++v) {
      int k0 = a_koff(v, hi);
      if (k0 < 16) {
        uint32_t lo = Pl[n16][k0];
        uint32_t h2 = Pl[n16][k0 + 1];
        pf.u[v] = lo | (h2 << 16);
      } else pf.u[v] = 0u;   // zero-padded K 16..31
    }

    // P(16x16) @ V(16x64): 4 output column tiles
#pragma unroll
    for (int c4 = 0; c4 < 4; ++c4) {
      Frag8 vf;
#pragma unroll
      for (int v = 0; v < 8; ++v) {
        int kk = a_koff(v, hi);
        if (kk >= 16) { vf.u[v] = 0u; continue; }   // K>=16 zero in A anyway
        int r0 = kt * 16 + kk, r1 = r0 + 1;
        int r0c = r0 < 65 ? r0 : 64, r1c = r1 < 65 ? r1 : 64;
        uint32_t lo = vb[base + (size_t)r0c * 1024 + c4 * 16 + n16];
        uint32_t hh = vb[base + (size_t)r1c * 1024 + c4 * 16 + n16];
        vf.u[v] = lo | (hh << 16);
      }
      oacc[c4] = __builtin_amdgcn_wmma_f32_16x16x32_bf16(
          false, pf.v, false, vf.v, (short)0, oacc[c4], false, false);
    }
    __syncthreads();
  }

#pragma unroll
  for (int e = 0; e < 8; ++e) {
    int t = qt * 16 + e + hi * 8;
    if (t >= 65) continue;
    float inv = 1.0f / lsum[e];
    u16* crow = ctxb + base + (size_t)t * 1024;
#pragma unroll
    for (int c4 = 0; c4 < 4; ++c4)
      crow[c4 * 16 + n16] = f2bf(oacc[c4][e] * inv);
  }
}

// ---------------------------------------------------------------------------
// Support kernels (VALU; small fraction of total work)
// ---------------------------------------------------------------------------
__global__ void cvt_kernel(const float* __restrict__ s, u16* __restrict__ d, long n) {
  long i  = (long)blockIdx.x * blockDim.x + threadIdx.x;
  long st = (long)gridDim.x * blockDim.x;
  for (; i < n; i += st) d[i] = f2bf(s[i]);
}

__global__ __launch_bounds__(256) void layernorm_kernel(
    const float* __restrict__ h, const float* __restrict__ g,
    const float* __restrict__ b, u16* __restrict__ out)
{
  const int row = blockIdx.x, tid = threadIdx.x;
  const int lane = tid & 31, wid = tid >> 5;
  __shared__ float red[8];
  const float* hr = h + (size_t)row * 1024;
  float x[4];
#pragma unroll
  for (int i = 0; i < 4; ++i) x[i] = hr[tid + i * 256];
  float s = x[0] + x[1] + x[2] + x[3];
#pragma unroll
  for (int m = 16; m >= 1; m >>= 1) s += __shfl_xor(s, m, 32);
  if (lane == 0) red[wid] = s;
  __syncthreads();
  float mu = 0.f;
#pragma unroll
  for (int i = 0; i < 8; ++i) mu += red[i];
  mu *= (1.0f / 1024.0f);
  __syncthreads();
  float vs = 0.f;
#pragma unroll
  for (int i = 0; i < 4; ++i) { float d = x[i] - mu; vs += d * d; }
#pragma unroll
  for (int m = 16; m >= 1; m >>= 1) vs += __shfl_xor(vs, m, 32);
  if (lane == 0) red[wid] = vs;
  __syncthreads();
  float var = 0.f;
#pragma unroll
  for (int i = 0; i < 8; ++i) var += red[i];
  var *= (1.0f / 1024.0f);
  float rs = rsqrtf(var + 1e-5f);
  u16* orow = out + (size_t)row * 1024;
#pragma unroll
  for (int i = 0; i < 4; ++i) {
    int col = tid + i * 256;
    orow[col] = f2bf((x[i] - mu) * rs * g[col] + b[col]);
  }
}

__global__ __launch_bounds__(256) void embed_kernel(
    const float* __restrict__ x, const float* __restrict__ pw,
    const float* __restrict__ pb, const float* __restrict__ apos,
    const float* __restrict__ femb, const float* __restrict__ remb,
    const float* __restrict__ demb, const float* __restrict__ ademb,
    float* __restrict__ h)
{
  const int bidx = blockIdx.x;
  const int b = bidx >> 6, s = bidx & 63;
  const int tid = threadIdx.x;
  __shared__ float xs[112];
  if (tid < 112) xs[tid] = x[((size_t)b * 112 + tid) * 64 + s];
  __syncthreads();
  const int rank = s >> 3, file = s & 7;
  const int diag = rank + file, adiag = rank - file + 7;
  float* hrow = h + (size_t)(b * 65 + s + 1) * 1024;
#pragma unroll
  for (int j = 0; j < 4; ++j) {
    int col = tid + j * 256;
    float acc = pb[col] + apos[s * 1024 + col] + femb[file * 1024 + col]
              + remb[rank * 1024 + col] + demb[diag * 1024 + col]
              + ademb[adiag * 1024 + col];
    for (int c = 0; c < 112; ++c) acc += xs[c] * pw[c * 1024 + col];
    hrow[col] = acc;
  }
}

__global__ __launch_bounds__(256) void cls_init_kernel(const float* __restrict__ cls,
                                                       float* __restrict__ h) {
  const int b = blockIdx.x, tid = threadIdx.x;
  float* hrow = h + (size_t)b * 65 * 1024;
#pragma unroll
  for (int j = 0; j < 4; ++j) hrow[tid + j * 256] = cls[tid + j * 256];
}

__global__ __launch_bounds__(256) void glu_kernel(const u16* __restrict__ p,
                                                  u16* __restrict__ g) {
  const size_t row = blockIdx.x;
  const int tid = threadIdx.x;
  const u16* pr = p + row * 8192;
  u16* gr = g + row * 4096;
#pragma unroll 4
  for (int j = 0; j < 16; ++j) {
    int col = tid + j * 256;
    float x1 = bf2f(pr[col]), x2 = bf2f(pr[col + 4096]);
    gr[col] = f2bf(gelu_exact(x1) * x2);
  }
}

__global__ void policy_gather_kernel(const float* __restrict__ ptmp,
                                     float* __restrict__ out) {
  const int bidx = blockIdx.x;            // b*64 + s
  const int b = bidx >> 6, s = bidx & 63;
  const int tid = threadIdx.x;
  if (tid < 72)
    out[(size_t)b * 4608 + s * 72 + tid] =
        ptmp[(size_t)(b * 65 + s + 1) * 72 + tid];
}

__global__ __launch_bounds__(256) void cls_gather_kernel(const u16* __restrict__ hn,
                                                         u16* __restrict__ clsb) {
  const int b = blockIdx.x, tid = threadIdx.x;
#pragma unroll
  for (int j = 0; j < 4; ++j)
    clsb[(size_t)b * 1024 + tid + j * 256] =
        hn[(size_t)b * 65 * 1024 + tid + j * 256];
}

__global__ __launch_bounds__(32) void wdl3_kernel(const u16* __restrict__ a,
                                                  const float* __restrict__ w3,
                                                  const float* __restrict__ b3,
                                                  float* __restrict__ out) {
  const int b = blockIdx.x, lane = threadIdx.x;
  if (lane < 3) {
    float acc = b3[lane];
    for (int k = 0; k < 256; ++k) acc += bf2f(a[b * 256 + k]) * w3[k * 3 + lane];
    out[b * 3 + lane] = acc;
  }
}

// ---------------------------------------------------------------------------
// Host orchestration
// ---------------------------------------------------------------------------
extern "C" void kernel_launch(void* const* d_in, const int* in_sizes, int n_in,
                              void* d_out, int out_size, void* d_ws, size_t ws_size,
                              hipStream_t stream)
{
  (void)in_sizes; (void)n_in; (void)out_size; (void)ws_size;
  const float* x         = (const float*)d_in[0];
  const float* proj_w    = (const float*)d_in[1];
  const float* proj_b    = (const float*)d_in[2];
  const float* cls       = (const float*)d_in[3];
  const float* abs_pos   = (const float*)d_in[4];
  const float* file_emb  = (const float*)d_in[5];
  const float* rank_emb  = (const float*)d_in[6];
  const float* diag_emb  = (const float*)d_in[7];
  const float* adiag_emb = (const float*)d_in[8];
  const float* ln1_g     = (const float*)d_in[9];
  const float* ln1_b     = (const float*)d_in[10];
  const float* wq        = (const float*)d_in[11];
  const float* bq        = (const float*)d_in[12];
  const float* wk        = (const float*)d_in[13];
  const float* bk        = (const float*)d_in[14];
  const float* wv        = (const float*)d_in[15];
  const float* bv        = (const float*)d_in[16];
  const float* wo        = (const float*)d_in[17];
  const float* bo        = (const float*)d_in[18];
  const float* bias_tab  = (const float*)d_in[19];
  const float* ln2_g     = (const float*)d_in[20];
  const float* ln2_b     = (const float*)d_in[21];
  const float* w1        = (const float*)d_in[22];
  const float* b1        = (const float*)d_in[23];
  const float* w2        = (const float*)d_in[24];
  const float* b2        = (const float*)d_in[25];
  const float* out_ln_g  = (const float*)d_in[26];
  const float* out_ln_b  = (const float*)d_in[27];
  const float* policy_w  = (const float*)d_in[28];
  const float* policy_b  = (const float*)d_in[29];
  const float* wdl_w1    = (const float*)d_in[30];
  const float* wdl_b1    = (const float*)d_in[31];
  const float* wdl_w2    = (const float*)d_in[32];
  const float* wdl_b2    = (const float*)d_in[33];
  const float* wdl_w3    = (const float*)d_in[34];
  const float* wdl_b3    = (const float*)d_in[35];
  float* outp = (float*)d_out;

  const long NTOK = 512L * 65;   // 33280 = 260 * 128

  // ---- workspace carve-out (256B aligned slabs) ----
  char* wsp = (char*)d_ws;
  auto take = [&](size_t bytes) {
    char* p = wsp; wsp += (bytes + 255) & ~(size_t)255; return p;
  };
  u16* wq_bf   = (u16*)take((size_t)15 * 1024 * 1024 * 2);
  u16* wk_bf   = (u16*)take((size_t)15 * 1024 * 1024 * 2);
  u16* wv_bf   = (u16*)take((size_t)15 * 1024 * 1024 * 2);
  u16* wo_bf   = (u16*)take((size_t)15 * 1024 * 1024 * 2);
  u16* w1_bf   = (u16*)take((size_t)15 * 1024 * 8192 * 2);
  u16* w2_bf   = (u16*)take((size_t)15 * 4096 * 1024 * 2);
  u16* polw_bf = (u16*)take((size_t)1024 * 72 * 2 + 256);
  u16* ww1_bf  = (u16*)take((size_t)1024 * 512 * 2);
  u16* ww2_bf  = (u16*)take((size_t)512 * 256 * 2);
  float* h     = (float*)take((size_t)NTOK * 1024 * 4);
  u16* hn      = (u16*)take((size_t)NTOK * 1024 * 2);
  u16* qb      = (u16*)take((size_t)NTOK * 1024 * 2);
  u16* kbuf    = (u16*)take((size_t)NTOK * 1024 * 2);
  u16* vbuf    = (u16*)take((size_t)NTOK * 1024 * 2);
  u16* ctxb    = (u16*)take((size_t)NTOK * 1024 * 2);
  u16* pbuf    = (u16*)take((size_t)NTOK * 8192 * 2);
  u16* gbuf    = (u16*)take((size_t)NTOK * 4096 * 2);
  float* ptmp  = (float*)take((size_t)NTOK * 72 * 4);
  u16* clsb    = (u16*)take((size_t)512 * 1024 * 2);
  u16* wdl1a   = (u16*)take((size_t)512 * 512 * 2);
  u16* wdl2a   = (u16*)take((size_t)512 * 256 * 2);

  auto cvt = [&](const float* s, u16* d, long n) {
    long g = (n + 2047) / 2048; if (g > 16384) g = 16384; if (g < 1) g = 1;
    cvt_kernel<<<dim3((unsigned)g), 256, 0, stream>>>(s, d, n);
  };

  // ---- weights fp32 -> bf16 (once per call; ~1GB read, ~45us of HBM) ----
  cvt(wq, wq_bf, 15L * 1024 * 1024);
  cvt(wk, wk_bf, 15L * 1024 * 1024);
  cvt(wv, wv_bf, 15L * 1024 * 1024);
  cvt(wo, wo_bf, 15L * 1024 * 1024);
  cvt(w1, w1_bf, 15L * 1024 * 8192);
  cvt(w2, w2_bf, 15L * 4096 * 1024);
  cvt(policy_w, polw_bf, 1024L * 72);
  cvt(wdl_w1, ww1_bf, 1024L * 512);
  cvt(wdl_w2, ww2_bf, 512L * 256);

  // ---- embeddings -> fp32 residual stream h ----
  cls_init_kernel<<<512, 256, 0, stream>>>(cls, h);
  embed_kernel<<<512 * 64, 256, 0, stream>>>(x, proj_w, proj_b, abs_pos,
                                             file_emb, rank_emb, diag_emb,
                                             adiag_emb, h);

  const dim3 gD(8, 260);    // N=1024 GEMMs over 33280 tokens
  const dim3 gF(64, 260);   // N=8192
  for (int l = 0; l < 15; ++l) {
    const size_t wD = (size_t)l * 1024 * 1024;
    layernorm_kernel<<<(unsigned)NTOK, 256, 0, stream>>>(h, ln1_g + l * 1024,
                                                         ln1_b + l * 1024, hn);
    gemm_bf16_kernel<EPI_BF16><<<gD, 256, 0, stream>>>(
        hn, wq_bf + wD, bq + l * 1024, qb, nullptr, (int)NTOK, 1024, 1024);
    gemm_bf16_kernel<EPI_BF16><<<gD, 256, 0, stream>>>(
        hn, wk_bf + wD, bk + l * 1024, kbuf, nullptr, (int)NTOK, 1024, 1024);
    gemm_bf16_kernel<EPI_BF16><<<gD, 256, 0, stream>>>(
        hn, wv_bf + wD, bv + l * 1024, vbuf, nullptr, (int)NTOK, 1024, 1024);
    attn_kernel<<<dim3(5, 16, 512), 32, 0, stream>>>(
        qb, kbuf, vbuf, bias_tab + (size_t)l * 15 * 15 * 16, ctxb);
    gemm_bf16_kernel<EPI_RESID><<<gD, 256, 0, stream>>>(
        ctxb, wo_bf + wD, bo + l * 1024, nullptr, h, (int)NTOK, 1024, 1024);
    layernorm_kernel<<<(unsigned)NTOK, 256, 0, stream>>>(h, ln2_g + l * 1024,
                                                         ln2_b + l * 1024, hn);
    gemm_bf16_kernel<EPI_BF16><<<gF, 256, 0, stream>>>(
        hn, w1_bf + (size_t)l * 1024 * 8192, b1 + l * 8192, pbuf, nullptr,
        (int)NTOK, 8192, 1024);
    glu_kernel<<<(unsigned)NTOK, 256, 0, stream>>>(pbuf, gbuf);
    gemm_bf16_kernel<EPI_RESID><<<gD, 256, 0, stream>>>(
        gbuf, w2_bf + (size_t)l * 4096 * 1024, b2 + l * 1024, nullptr, h,
        (int)NTOK, 1024, 4096);
  }

  // ---- heads ----
  layernorm_kernel<<<(unsigned)NTOK, 256, 0, stream>>>(h, out_ln_g, out_ln_b, hn);
  gemm_bf16_kernel<EPI_F32><<<dim3(1, 260), 256, 0, stream>>>(
      hn, polw_bf, policy_b, ptmp, nullptr, (int)NTOK, 72, 1024);
  policy_gather_kernel<<<512 * 64, 128, 0, stream>>>(ptmp, outp);
  cls_gather_kernel<<<512, 256, 0, stream>>>(hn, clsb);
  gemm_bf16_kernel<EPI_BF16_GELU><<<dim3(4, 4), 256, 0, stream>>>(
      clsb, ww1_bf, wdl_b1, wdl1a, nullptr, 512, 512, 1024);
  gemm_bf16_kernel<EPI_BF16_GELU><<<dim3(2, 4), 256, 0, stream>>>(
      wdl1a, ww2_bf, wdl_b2, wdl2a, nullptr, 512, 256, 512);
  wdl3_kernel<<<512, 32, 0, stream>>>(wdl2a, wdl_w3, wdl_b3,
                                      outp + (size_t)512 * 4608);
}
